// SelfAttention_8091718386488
// MI455X (gfx1250) — matrix operand order repaired and checked by reading, not compiled, not run on hardware
//
#include <hip/hip_runtime.h>
#include <hip/hip_bf16.h>

#define BATCH 4
#define SEQ   2048
#define DMODEL 1024
#define HEADS 16
#define HDIM  64

typedef __attribute__((ext_vector_type(16))) __bf16 v16bf;
typedef __attribute__((ext_vector_type(8)))  __bf16 v8bf;
typedef __attribute__((ext_vector_type(8)))  float  v8f;
typedef __attribute__((ext_vector_type(4)))  unsigned int u32x4;
typedef __attribute__((ext_vector_type(8)))  int i32x8;
typedef __attribute__((ext_vector_type(4)))  int i32x4;

#if __has_include(<hip/amd_detail/amd_gfx1250_TDM.h>)
#define TDM_6ARG 1
#else
#define TDM_6ARG 0
#endif

// ---------------------------------------------------------------------------
// fp32 -> bf16 conversion
// ---------------------------------------------------------------------------
__global__ void __launch_bounds__(256) cvt_f32_bf16(const float* __restrict__ in,
                                                    __bf16* __restrict__ out, int n) {
    int i = blockIdx.x * 256 + threadIdx.x;
    if (i < n) out[i] = (__bf16)in[i];
}

// ---------------------------------------------------------------------------
// Fragment loaders for 16-bit WMMA operands, contiguous k order used by both A and B.
// A (16x32, MxK): lane m (m<16) holds row m, K = kb+0..15;
//                 lane m+16 holds row m, K = kb+16..31.
// B (32x16, KxN): lane n (n<16) holds col n, K = kb+0..15;
//                 lane n+16 holds col n, K = kb+16..31.
// Both A rows and B columns are contiguous (K-major) in our memory layouts.
// ---------------------------------------------------------------------------
__device__ inline v16bf load_frag_a(const __bf16* base, int ld, int row, int kb, int lane) {
    const __bf16* p = base + (size_t)(row + (lane & 15)) * ld + kb + ((lane >> 4) << 4);
    v8bf lo = *(const v8bf*)p;
    v8bf hi = *(const v8bf*)(p + 8);
    v16bf f;
#pragma unroll
    for (int i = 0; i < 8; ++i) { f[i] = lo[i]; f[i + 8] = hi[i]; }
    return f;
}

__device__ inline v16bf load_frag_b(const __bf16* base, int ld, int col, int kb, int lane) {
    const __bf16* p = base + (size_t)(col + (lane & 15)) * ld + kb + ((lane >> 4) << 4);
    v8bf lo = *(const v8bf*)p;
    v8bf hi = *(const v8bf*)(p + 8);
    v16bf f;
#pragma unroll
    for (int i = 0; i < 8; ++i) { f[i] = lo[i]; f[i + 8] = hi[i]; }
    return f;
}

// B fragment from an LDS tile laid out [64 cols][32 k] row-major (bf16)
__device__ inline v16bf load_frag_b_lds(const __bf16* tile, int colLocal, int lane) {
    const __bf16* p = tile + (size_t)(colLocal + (lane & 15)) * 32 + ((lane >> 4) << 4);
    v8bf lo = *(const v8bf*)p;
    v8bf hi = *(const v8bf*)(p + 8);
    v16bf f;
#pragma unroll
    for (int i = 0; i < 8; ++i) { f[i] = lo[i]; f[i + 8] = hi[i]; }
    return f;
}

__device__ inline v8f wmma_bf16(v16bf a, v16bf b, v8f c) {
    return __builtin_amdgcn_wmma_f32_16x16x32_bf16(false, a, false, b, (short)0, c,
                                                   false, false);
}

// ---------------------------------------------------------------------------
// TDM: DMA a 64(rows) x 32(k) bf16 tile of W into LDS.
// D# per CDNA5 ISA 8.3/8.4: 2-D tensor, dim0 = K (contiguous), dim1 = rows,
// tensor_dim0_stride = K, tile = 32 x 64, data_size = 2B, type = 2.
// ---------------------------------------------------------------------------
__device__ inline void tdm_load_w_tile(const __bf16* W, int K, int nrows,
                                       int nbase, int kb, __bf16* ldsDst) {
    unsigned long long ga =
        (unsigned long long)(uintptr_t)(W + (size_t)nbase * K + kb);
    unsigned int lds = (unsigned int)(uintptr_t)ldsDst;  // LDS offset = addr[31:0]

    u32x4 g0;
    g0[0] = 1u;                                             // count=1 (valid)
    g0[1] = lds;                                            // lds_addr (bytes)
    g0[2] = (unsigned int)ga;                               // global_addr[31:0]
    g0[3] = (unsigned int)((ga >> 32) & 0x01FFFFFFu) | (2u << 30);  // [56:32]+type=2

    unsigned int d0 = (unsigned int)K;       // tensor_dim0
    unsigned int d1 = (unsigned int)nrows;   // tensor_dim1
    i32x8 g1;
    g1[0] = (int)(1u << 16);                                // data_size=1 -> 2 bytes
    g1[1] = (int)((d0 & 0xFFFFu) << 16);                    // tensor_dim0[15:0]
    g1[2] = (int)((d0 >> 16) | ((d1 & 0xFFFFu) << 16));     // dim0 hi | dim1 lo
    g1[3] = (int)((d1 >> 16) | (32u << 16));                // dim1 hi | tile_dim0=32
    g1[4] = (int)(64u);                                     // tile_dim1=64, tile_dim2=0
    g1[5] = (int)((unsigned int)K);                         // dim0_stride low32
    g1[6] = 0;                                              // stride hi / dim1_stride
    g1[7] = 0;

    i32x4 gz;
    gz[0] = 0; gz[1] = 0; gz[2] = 0; gz[3] = 0;
#if TDM_6ARG
    i32x8 z8;
#pragma unroll
    for (int i = 0; i < 8; ++i) z8[i] = 0;
    __builtin_amdgcn_tensor_load_to_lds(g0, g1, gz, gz, z8, 0);
#else
    __builtin_amdgcn_tensor_load_to_lds(g0, g1, gz, gz, 0);
#endif
}

// ---------------------------------------------------------------------------
// GEMM: C[M,N] = A[M,K] @ W[N,K]^T   (A, W bf16 K-major rows)
// One wave computes a 16(M) x 64(N) tile. Block = 4 waves = 64x64 tile.
// W tile (64x32) is TDM-DMA'd into LDS once per block per k-step, double
// buffered and tracked with TENSORcnt; A fragments stream from global.
// VARIANT 0: bf16 out, head-major   [b,h,s,hd]   (Q, K)
// VARIANT 1: bf16 out, head-major T [b,h,hd,s]   (V transposed for PV GEMM)
// VARIANT 2: f32  out, row-major    [m,n]        (final projection)
// ---------------------------------------------------------------------------
template <int VARIANT>
__global__ void __launch_bounds__(128) proj_gemm(const __bf16* __restrict__ A,
                                                 const __bf16* __restrict__ W,
                                                 void* __restrict__ out,
                                                 int M, int N, int K) {
    __shared__ __bf16 tileW[2][64 * 32];

    const int lane = threadIdx.x & 31;
    const int wave = threadIdx.x >> 5;
    const int mbase = blockIdx.x * 64 + wave * 16;
    const int nbase = blockIdx.y * 64;

    v8f acc[4];
#pragma unroll
    for (int t = 0; t < 4; ++t)
#pragma unroll
        for (int i = 0; i < 8; ++i) acc[t][i] = 0.0f;

    const int nsteps = K >> 5;

    if (wave == 0) tdm_load_w_tile(W, K, N, nbase, 0, &tileW[0][0]);

    for (int it = 0; it < nsteps; ++it) {
        const int kb = it << 5;
        if (it + 1 < nsteps) {
            if (wave == 0) {
                tdm_load_w_tile(W, K, N, nbase, kb + 32, &tileW[(it + 1) & 1][0]);
                __builtin_amdgcn_s_wait_tensorcnt(1);   // current tile done
            }
        } else {
            if (wave == 0) __builtin_amdgcn_s_wait_tensorcnt(0);
        }
        __syncthreads();  // tile visible to all 4 waves

        const __bf16* tw = &tileW[it & 1][0];
        v16bf af = load_frag_a(A, K, mbase, kb, lane);
#pragma unroll
        for (int t = 0; t < 4; ++t) {
            v16bf bf = load_frag_b_lds(tw, 16 * t, lane);
            acc[t] = wmma_bf16(af, bf, acc[t]);
        }
        __syncthreads();  // all waves done reading before next overwrite
    }

    // C layout: VGPR i, lanes 0-15 -> row mbase+i, lanes 16-31 -> row mbase+8+i
#pragma unroll
    for (int t = 0; t < 4; ++t) {
#pragma unroll
        for (int i = 0; i < 8; ++i) {
            int m = mbase + i + ((lane >> 4) << 3);
            int n = nbase + t * 16 + (lane & 15);
            float v = acc[t][i];
            if (VARIANT == 2) {
                ((float*)out)[(size_t)m * N + n] = v;
            } else {
                int b = m >> 11, s = m & (SEQ - 1);
                int h = n >> 6, hd = n & (HDIM - 1);
                __bf16* o = (__bf16*)out;
                if (VARIANT == 0)
                    o[(((size_t)(b * HEADS + h)) * SEQ + s) * HDIM + hd] = (__bf16)v;
                else
                    o[(((size_t)(b * HEADS + h)) * HDIM + hd) * SEQ + s] = (__bf16)v;
            }
        }
    }
}

// ---------------------------------------------------------------------------
// Flash attention, causal. Grid: (SEQ/64, BATCH*HEADS), block = 128 (4 waves).
// Each wave: 16 query rows x hd=64, online softmax over key chunks of 32.
// ---------------------------------------------------------------------------
__global__ void __launch_bounds__(128) attn_kernel(const __bf16* __restrict__ Qh,
                                                   const __bf16* __restrict__ Kh,
                                                   const __bf16* __restrict__ Vt,
                                                   __bf16* __restrict__ Ctx) {
    __shared__ __bf16 pbuf[4 * 16 * 32];

    const int lane = threadIdx.x & 31;
    const int wave = threadIdx.x >> 5;
    const int bh = blockIdx.y;               // b*HEADS + h
    const int qbase = blockIdx.x * 64 + wave * 16;

    const __bf16* q = Qh + (size_t)bh * SEQ * HDIM;
    const __bf16* k = Kh + (size_t)bh * SEQ * HDIM;
    const __bf16* vt = Vt + (size_t)bh * HDIM * SEQ;
    __bf16* pb = pbuf + wave * 16 * 32;

    // Q fragments for this wave's 16 rows (hd split 0..31 / 32..63)
    v16bf qf[2];
    qf[0] = load_frag_a(q, HDIM, qbase, 0, lane);
    qf[1] = load_frag_a(q, HDIM, qbase, 32, lane);

    // softmax state per row (VGPR i <-> row qbase + i + 8*(lane>=16))
    float mstat[8], lstat[8];
    v8f o[4];
#pragma unroll
    for (int i = 0; i < 8; ++i) { mstat[i] = -1e30f; lstat[i] = 0.0f; }
#pragma unroll
    for (int t = 0; t < 4; ++t)
#pragma unroll
        for (int i = 0; i < 8; ++i) o[t][i] = 0.0f;

    const float SC = 0.125f * 1.44269504088896340736f;  // 1/sqrt(64) * log2(e)
    const int half8 = (lane >> 4) << 3;

    for (int jb = 0; jb < qbase + 16; jb += 32) {
        // ---- S = Q @ K^T for 16 q-rows x 32 keys ----
        v8f sacc[2];
#pragma unroll
        for (int t = 0; t < 2; ++t) {
#pragma unroll
            for (int i = 0; i < 8; ++i) sacc[t][i] = 0.0f;
#pragma unroll
            for (int c = 0; c < 2; ++c) {
                v16bf kf = load_frag_b(k, HDIM, jb + 16 * t, 32 * c, lane);
                sacc[t] = wmma_bf16(qf[c], kf, sacc[t]);
            }
        }

        // ---- scale to log2 domain + causal mask ----
#pragma unroll
        for (int t = 0; t < 2; ++t)
#pragma unroll
            for (int i = 0; i < 8; ++i) {
                float tv = sacc[t][i] * SC;
                int key = jb + t * 16 + (lane & 15);
                int qrow = qbase + i + half8;
                sacc[t][i] = (key > qrow) ? -1e30f : tv;
            }

        // ---- row max (16-lane half reduction) + online rescale ----
        float sc[8];
#pragma unroll
        for (int i = 0; i < 8; ++i) {
            float rm = fmaxf(sacc[0][i], sacc[1][i]);
#pragma unroll
            for (int d = 1; d < 16; d <<= 1) rm = fmaxf(rm, __shfl_xor(rm, d));
            float nm = fmaxf(mstat[i], rm);
            sc[i] = __builtin_exp2f(mstat[i] - nm);
            mstat[i] = nm;
        }

        // ---- p = 2^(s - m), row sum, l update, O rescale ----
#pragma unroll
        for (int i = 0; i < 8; ++i) {
            float p0 = __builtin_exp2f(sacc[0][i] - mstat[i]);
            float p1 = __builtin_exp2f(sacc[1][i] - mstat[i]);
            sacc[0][i] = p0;
            sacc[1][i] = p1;
            float rs = p0 + p1;
#pragma unroll
            for (int d = 1; d < 16; d <<= 1) rs += __shfl_xor(rs, d);
            lstat[i] = lstat[i] * sc[i] + rs;
        }
#pragma unroll
        for (int t = 0; t < 4; ++t)
#pragma unroll
            for (int i = 0; i < 8; ++i) o[t][i] *= sc[i];

        // ---- transpose P (C-layout) to A-layout via LDS ----
#pragma unroll
        for (int t = 0; t < 2; ++t)
#pragma unroll
            for (int i = 0; i < 8; ++i) {
                int row = i + half8;
                int col = t * 16 + (lane & 15);
                pb[row * 32 + col] = (__bf16)sacc[t][i];
            }
        {
            const __bf16* pr = pb + (lane & 15) * 32 + ((lane >> 4) << 4);
            v8bf lo = *(const v8bf*)pr;
            v8bf hi = *(const v8bf*)(pr + 8);
            v16bf pf;
#pragma unroll
            for (int i = 0; i < 8; ++i) { pf[i] = lo[i]; pf[i + 8] = hi[i]; }

            // ---- O += P @ V  (V transposed: rows = hd, contiguous over keys) ----
#pragma unroll
            for (int t = 0; t < 4; ++t) {
                const __bf16* vp =
                    vt + (size_t)(t * 16 + (lane & 15)) * SEQ + jb + ((lane >> 4) << 4);
                v8bf vlo = *(const v8bf*)vp;
                v8bf vhi = *(const v8bf*)(vp + 8);
                v16bf vf;
#pragma unroll
                for (int i = 0; i < 8; ++i) { vf[i] = vlo[i]; vf[i + 8] = vhi[i]; }
                o[t] = wmma_bf16(pf, vf, o[t]);
            }
        }
    }

    // ---- normalize and store ctx as bf16 [b, s, h*64+hd] for final GEMM ----
    const int b = bh >> 4, h = bh & 15;
#pragma unroll
    for (int i = 0; i < 8; ++i) {
        float inv = 1.0f / lstat[i];
        int s = qbase + i + half8;
#pragma unroll
        for (int t = 0; t < 4; ++t) {
            int col = h * HDIM + t * 16 + (lane & 15);
            Ctx[((size_t)b * SEQ + s) * DMODEL + col] = (__bf16)(o[t][i] * inv);
        }
    }
}

// ---------------------------------------------------------------------------
// Launcher
// ---------------------------------------------------------------------------
extern "C" void kernel_launch(void* const* d_in, const int* in_sizes, int n_in,
                              void* d_out, int out_size, void* d_ws, size_t ws_size,
                              hipStream_t stream) {
    const float* x  = (const float*)d_in[0];
    const float* wq = (const float*)d_in[1];
    const float* wk = (const float*)d_in[2];
    const float* wv = (const float*)d_in[3];
    const float* wo = (const float*)d_in[4];

    const size_t NX = (size_t)BATCH * SEQ * DMODEL;   // 8,388,608
    const size_t NW = (size_t)DMODEL * DMODEL;        // 1,048,576

    __bf16* ws = (__bf16*)d_ws;
    __bf16* Xb  = ws;                 // NX
    __bf16* Wqb = Xb + NX;            // NW
    __bf16* Wkb = Wqb + NW;
    __bf16* Wvb = Wkb + NW;
    __bf16* Wob = Wvb + NW;
    __bf16* Qh  = Wob + NW;           // NX  [b,h,s,hd]
    __bf16* Kh  = Qh + NX;            // NX  [b,h,s,hd]
    __bf16* Vt  = Kh + NX;            // NX  [b,h,hd,s]
    __bf16* Ctx = Vt + NX;            // NX  [b,s,d]

    // fp32 -> bf16
    cvt_f32_bf16<<<(int)(NX / 256), 256, 0, stream>>>(x, Xb, (int)NX);
    cvt_f32_bf16<<<(int)(NW / 256), 256, 0, stream>>>(wq, Wqb, (int)NW);
    cvt_f32_bf16<<<(int)(NW / 256), 256, 0, stream>>>(wk, Wkb, (int)NW);
    cvt_f32_bf16<<<(int)(NW / 256), 256, 0, stream>>>(wv, Wvb, (int)NW);
    cvt_f32_bf16<<<(int)(NW / 256), 256, 0, stream>>>(wo, Wob, (int)NW);

    const int M = BATCH * SEQ;  // 8192
    dim3 gGemm(M / 64, DMODEL / 64);
    proj_gemm<0><<<gGemm, 128, 0, stream>>>(Xb, Wqb, Qh, M, DMODEL, DMODEL);
    proj_gemm<0><<<gGemm, 128, 0, stream>>>(Xb, Wkb, Kh, M, DMODEL, DMODEL);
    proj_gemm<1><<<gGemm, 128, 0, stream>>>(Xb, Wvb, Vt, M, DMODEL, DMODEL);

    dim3 gAttn(SEQ / 64, BATCH * HEADS);
    attn_kernel<<<gAttn, 128, 0, stream>>>(Qh, Kh, Vt, Ctx);

    proj_gemm<2><<<gGemm, 128, 0, stream>>>(Ctx, Wob, d_out, M, DMODEL, DMODEL);
}
